// Dilated_res_block_27066883899418
// MI455X (gfx1250) — matrix operand order repaired and checked by reading, not compiled, not run on hardware
//
#include <hip/hip_runtime.h>
#include <hip/hip_bf16.h>

typedef __attribute__((ext_vector_type(16))) _Float16 v16h;
typedef __attribute__((ext_vector_type(8)))  float    v8f;

#define DEV __device__ __forceinline__

constexpr int   BATCH = 2;
constexpr int   NPTS  = 16384;
constexpr int   KNBR  = 16;
constexpr int   CIN   = 64;
constexpr int   D2    = 64;
constexpr int   DOUT  = 128;
constexpr int   DO2   = 256;
constexpr float BNINV = 0.9999950000374997f; // 1/sqrt(1+1e-5)

// ---- intra-wave LDS write -> cross-lane read ordering -----------------------
DEV void wave_lds_fence() {
  __builtin_amdgcn_wave_barrier();
  asm volatile("s_wait_dscnt 0x0" ::: "memory");
  __builtin_amdgcn_wave_barrier();
}

// ---- WMMA fragment loaders (CDNA5 16x16x32 f16 layouts) ---------------------
// A (16x32 f16): lane L -> row M = L&15; K = 16*(L>=16) + i.
DEV v16h load_a_h(const _Float16* base, int ld, int row0, int kc, int ln) {
  const _Float16* r = base + (size_t)(row0 + (ln & 15)) * ld + kc + ((ln >> 4) ? 16 : 0);
  v16h a;
#pragma unroll
  for (int i = 0; i < 16; ++i) a[i] = r[i];
  return a;
}

DEV v16h load_a_f(const float* base, int ld, int row0, int kc, int ln) {
  const float* r = base + (size_t)(row0 + (ln & 15)) * ld + kc + ((ln >> 4) ? 16 : 0);
  v16h a;
#pragma unroll
  for (int i = 0; i < 16; ++i) a[i] = (_Float16)r[i];
  return a;
}

// B (32x16 f16): lane L -> col N = L&15; K = 16*(L>=16) + i.
// BN gamma * 1/sqrt(1+eps) folded into the fragment (g == nullptr -> raw fc).
DEV v16h load_b_full(const float* W, int ld, int obase, int kc,
                     const float* g, float inv, int ln) {
  int o = obase + (ln & 15);
  float s = (g ? g[o] : 1.0f) * inv;
  const float* r = W + (size_t)o * ld + kc + ((ln >> 4) ? 16 : 0);
  v16h b;
#pragma unroll
  for (int i = 0; i < 16; ++i) b[i] = (_Float16)(r[i] * s);
  return b;
}

DEV v16h load_b_pad(const float* W, int ld, int kvalid, int obase, int kc,
                    const float* g, float inv, int ln) {
  int o = obase + (ln & 15);
  float s = (g ? g[o] : 1.0f) * inv;
  int k0 = kc + ((ln >> 4) ? 16 : 0);
  v16h b;
#pragma unroll
  for (int i = 0; i < 16; ++i) {
    int k = k0 + i;
    b[i] = (_Float16)((k < kvalid) ? W[(size_t)o * ld + k] * s : 0.0f);
  }
  return b;
}

DEV v8f wmma32(v16h a, v16h b, v8f c) {
  return __builtin_amdgcn_wmma_f32_16x16x32_f16(false, a, false, b, (short)0, c,
                                                false, false);
}

// ============================================================================
// Kernel 1: mlp1  (f_in [B,64,N] -> f1 [B,N,64], BN+ReLU)
// ============================================================================
__global__ void __launch_bounds__(256)
k_mlp1(const float* __restrict__ feature, const float* __restrict__ w,
       const float* __restrict__ g, const float* __restrict__ bia,
       float* __restrict__ f1) {
  __shared__ _Float16 sha[128 * 64];
  const int bid = blockIdx.x;
  const int b   = bid / (NPTS / 128);
  const int n0  = (bid % (NPTS / 128)) * 128;
  const int t   = threadIdx.x;
  // coalesced staging: each thread loads float4 along n for a fixed channel
#pragma unroll
  for (int i = 0; i < 8; ++i) {
    int q = i * 256 + t;           // 2048 float4 total
    int c = q >> 5, p4 = q & 31;   // c in [0,64), p4 in [0,32)
    float4 v = *(const float4*)(feature + (size_t)(b * CIN + c) * NPTS + n0 + p4 * 4);
    sha[(p4 * 4 + 0) * 64 + c] = (_Float16)v.x;
    sha[(p4 * 4 + 1) * 64 + c] = (_Float16)v.y;
    sha[(p4 * 4 + 2) * 64 + c] = (_Float16)v.z;
    sha[(p4 * 4 + 3) * 64 + c] = (_Float16)v.w;
  }
  __syncthreads();
  const int wv = t >> 5, ln = t & 31;
  const int col = ln & 15, hi = ln >> 4;
  const int nt = wv & 3, ob = nt * 16, o = ob + col;
  // hoisted per-wave B fragments + bias
  const v16h bf0 = load_b_full(w, 64, ob, 0, g, BNINV, ln);
  const v16h bf1 = load_b_full(w, 64, ob, 32, g, BNINV, ln);
  const float bias = bia[o];
#pragma unroll
  for (int i = 0; i < 4; ++i) {
    int mt = (wv >> 2) + 2 * i;
    v8f acc = {};
    acc = wmma32(load_a_h(sha, 64, mt * 16, 0, ln), bf0, acc);
    acc = wmma32(load_a_h(sha, 64, mt * 16, 32, ln), bf1, acc);
    float* dst = f1 + (size_t)(b * NPTS + n0 + mt * 16 + 8 * hi) * D2 + o;
#pragma unroll
    for (int j = 0; j < 8; ++j) dst[j * D2] = fmaxf(acc[j] + bias, 0.0f);
  }
}

// ============================================================================
// Kernel 2: rel-pos enc (bb1) + att_pool1 -> agg1 [B,N,64]
// One WG = 16 points, 256 threads = 8 waves, dynamic LDS (99584 B).
// ============================================================================
__global__ void __launch_bounds__(256)
k_block1(const float* __restrict__ xyz, const int* __restrict__ nidx,
         const float* __restrict__ f1,
         const float* __restrict__ bb1w, const float* __restrict__ bb1g,
         const float* __restrict__ bb1b, const float* __restrict__ a1fc,
         const float* __restrict__ a1w, const float* __restrict__ a1g,
         const float* __restrict__ a1b, float* __restrict__ agg1) {
  extern __shared__ char smem[];
  _Float16* in10 = (_Float16*)(smem);               // [256][32]  16384 B
  _Float16* catm = (_Float16*)(smem + 16384);       // [256][128] 65536 B
  float*    aggs = (float*)(smem + 81920);          // [16][128]   8192 B
  float*    atts = (float*)(smem + 90112);          // [8][256]    8192 B
  int*      sidx = (int*)(smem + 98304);            // [256]       1024 B
  float*    pxyz = (float*)(smem + 99328);          // [16][4]      256 B

  const int bid = blockIdx.x;
  const int b   = bid >> 10;
  const int n0  = (bid & 1023) * 16;
  const int t   = threadIdx.x;
  const int wv  = t >> 5, ln = t & 31;
  const int col = ln & 15, hi = ln >> 4;

  sidx[t] = nidx[(size_t)(b * NPTS + n0 + (t >> 4)) * KNBR + (t & 15)];
  if (t < 48) {
    int p = t / 3, d = t % 3;
    pxyz[p * 4 + d] = xyz[(size_t)(b * NPTS + n0 + p) * 3 + d];
  }
  __syncthreads();

  { // padded 10-ch rel-pos rows + gather f_nb (f32->f16) into cat[:,0:64]
    int idn = sidx[t];
    const float* nx = xyz + (size_t)(b * NPTS + idn) * 3;
    int p = t >> 4;
    float tx = pxyz[p * 4 + 0], ty = pxyz[p * 4 + 1], tz = pxyz[p * 4 + 2];
    float ax = nx[0], ay = nx[1], az = nx[2];
    float rx = tx - ax, ry = ty - ay, rz = tz - az;
    float dis = sqrtf(rx * rx + ry * ry + rz * rz);
    _Float16* row = in10 + t * 32;
    row[0] = (_Float16)dis; row[1] = (_Float16)rx; row[2] = (_Float16)ry;
    row[3] = (_Float16)rz;  row[4] = (_Float16)tx; row[5] = (_Float16)ty;
    row[6] = (_Float16)tz;  row[7] = (_Float16)ax; row[8] = (_Float16)ay;
    row[9] = (_Float16)az;
#pragma unroll
    for (int j = 10; j < 32; ++j) row[j] = (_Float16)0.0f;
    const float4* src = (const float4*)(f1 + (size_t)(b * NPTS + idn) * D2);
    _Float16* dst = catm + t * 128;
#pragma unroll
    for (int c4 = 0; c4 < 16; ++c4) {
      float4 v = src[c4];
      dst[c4 * 4 + 0] = (_Float16)v.x;
      dst[c4 * 4 + 1] = (_Float16)v.y;
      dst[c4 * 4 + 2] = (_Float16)v.z;
      dst[c4 * 4 + 3] = (_Float16)v.w;
    }
  }
  __syncthreads();

  // bb1: 10->64 (K padded) -> cat[:,64:128], BN+ReLU. nt invariant per wave.
  {
    const int nt = wv & 3, ob = nt * 16, o = ob + col;
    const v16h bfb = load_b_pad(bb1w, 10, 10, ob, 0, bb1g, BNINV, ln);
    const float bias = bb1b[o];
#pragma unroll
    for (int i = 0; i < 8; ++i) {
      int mt = (wv >> 2) + 2 * i;
      v8f acc = {};
      acc = wmma32(load_a_h(in10, 32, mt * 16, 0, ln), bfb, acc);
      _Float16* dst = catm + (mt * 16 + 8 * hi) * 128 + 64 + o;
#pragma unroll
      for (int j = 0; j < 8; ++j)
        dst[j * 128] = (_Float16)fmaxf(acc[j] + bias, 0.0f);
    }
  }
  __syncthreads();

  // att1: per wave a fixed 16-ch tile (nt == wv); loop over points p
  {
    const int ob = wv * 16;
    const v16h w0 = load_b_full(a1fc, 128, ob, 0, nullptr, 1.0f, ln);
    const v16h w1 = load_b_full(a1fc, 128, ob, 32, nullptr, 1.0f, ln);
    const v16h w2 = load_b_full(a1fc, 128, ob, 64, nullptr, 1.0f, ln);
    const v16h w3 = load_b_full(a1fc, 128, ob, 96, nullptr, 1.0f, ln);
    float* sc = atts + wv * 256;
#pragma unroll 1
    for (int p = 0; p < 16; ++p) {
      v8f acc = {};
      acc = wmma32(load_a_h(catm, 128, p * 16, 0, ln), w0, acc);
      acc = wmma32(load_a_h(catm, 128, p * 16, 32, ln), w1, acc);
      acc = wmma32(load_a_h(catm, 128, p * 16, 64, ln), w2, acc);
      acc = wmma32(load_a_h(catm, 128, p * 16, 96, ln), w3, acc);
#pragma unroll
      for (int j = 0; j < 8; ++j) sc[(j + 8 * hi) * 16 + col] = acc[j];
      wave_lds_fence();
      if (ln < 16) {
        int o = ob + ln;
        float m = -1e30f;
#pragma unroll
        for (int k = 0; k < 16; ++k) m = fmaxf(m, sc[k * 16 + ln]);
        float s = 0.f, tt = 0.f;
#pragma unroll
        for (int k = 0; k < 16; ++k) {
          float e = __expf(sc[k * 16 + ln] - m);
          s += e;
          tt += e * (float)catm[(p * 16 + k) * 128 + o];
        }
        aggs[p * 128 + o] = tt / s;
      }
      wave_lds_fence();
    }
  }
  __syncthreads();

  // a1: 128 -> 64, BN+ReLU -> agg1
  if (wv < 4) {
    int ob = wv * 16, o = ob + col;
    v8f acc = {};
#pragma unroll
    for (int kc = 0; kc < 128; kc += 32) {
      v16h af = load_a_f(aggs, 128, 0, kc, ln);
      v16h bf = load_b_full(a1w, 128, ob, kc, a1g, BNINV, ln);
      acc = wmma32(af, bf, acc);
    }
    float bias = a1b[o];
    float* dst = agg1 + (size_t)(b * NPTS + n0 + 8 * hi) * D2 + o;
#pragma unroll
    for (int j = 0; j < 8; ++j) dst[j * D2] = fmaxf(acc[j] + bias, 0.0f);
  }
}

// ============================================================================
// Kernel 3: recompute f_xyz, bb2, att_pool2, a2, fused m2+shortcut+leaky -> out
// Dynamic LDS 138496 B.
// ============================================================================
__global__ void __launch_bounds__(256)
k_block2(const float* __restrict__ feature, const float* __restrict__ xyz,
         const int* __restrict__ nidx, const float* __restrict__ agg1,
         const float* __restrict__ bb1w, const float* __restrict__ bb1g,
         const float* __restrict__ bb1b, const float* __restrict__ bb2w,
         const float* __restrict__ bb2g, const float* __restrict__ bb2b,
         const float* __restrict__ a2fc, const float* __restrict__ a2w,
         const float* __restrict__ a2g, const float* __restrict__ a2b,
         const float* __restrict__ m2w, const float* __restrict__ m2g,
         const float* __restrict__ m2b, const float* __restrict__ scw,
         const float* __restrict__ scg, const float* __restrict__ scb,
         float* __restrict__ out) {
  extern __shared__ char smem[];
  _Float16* in10 = (_Float16*)(smem);                // [256][32]  16384
  _Float16* fxyz = (_Float16*)(smem + 16384);        // [256][64]  32768
  _Float16* catm = (_Float16*)(smem + 49152);        // [256][128] 65536
  float*    aggs = (float*)(smem + 114688);          // [16][128]   8192
  float*    atts = (float*)(smem + 122880);          // [8][256]    8192
  _Float16* finh = (_Float16*)(smem + 131072);       // [16][64]    2048
  _Float16* f2h  = (_Float16*)(smem + 133120);       // [16][128]   4096
  int*      sidx = (int*)(smem + 137216);            // [256]       1024
  float*    pxyz = (float*)(smem + 138240);          // [16][4]      256

  const int bid = blockIdx.x;
  const int b   = bid >> 10;
  const int n0  = (bid & 1023) * 16;
  const int t   = threadIdx.x;
  const int wv  = t >> 5, ln = t & 31;
  const int col = ln & 15, hi = ln >> 4;

  sidx[t] = nidx[(size_t)(b * NPTS + n0 + (t >> 4)) * KNBR + (t & 15)];
  if (t < 48) {
    int p = t / 3, d = t % 3;
    pxyz[p * 4 + d] = xyz[(size_t)(b * NPTS + n0 + p) * 3 + d];
  }
  // stage f_in tile [16][64] f16 (coalesced float4 along n)
  {
    int c = t >> 2, p4 = t & 3;
    float4 v = *(const float4*)(feature + (size_t)(b * CIN + c) * NPTS + n0 + p4 * 4);
    finh[(p4 * 4 + 0) * 64 + c] = (_Float16)v.x;
    finh[(p4 * 4 + 1) * 64 + c] = (_Float16)v.y;
    finh[(p4 * 4 + 2) * 64 + c] = (_Float16)v.z;
    finh[(p4 * 4 + 3) * 64 + c] = (_Float16)v.w;
  }
  __syncthreads();

  { // rel-pos rows + gather agg1 neighbors into cat[:,0:64]
    int idn = sidx[t];
    const float* nx = xyz + (size_t)(b * NPTS + idn) * 3;
    int p = t >> 4;
    float tx = pxyz[p * 4 + 0], ty = pxyz[p * 4 + 1], tz = pxyz[p * 4 + 2];
    float ax = nx[0], ay = nx[1], az = nx[2];
    float rx = tx - ax, ry = ty - ay, rz = tz - az;
    float dis = sqrtf(rx * rx + ry * ry + rz * rz);
    _Float16* row = in10 + t * 32;
    row[0] = (_Float16)dis; row[1] = (_Float16)rx; row[2] = (_Float16)ry;
    row[3] = (_Float16)rz;  row[4] = (_Float16)tx; row[5] = (_Float16)ty;
    row[6] = (_Float16)tz;  row[7] = (_Float16)ax; row[8] = (_Float16)ay;
    row[9] = (_Float16)az;
#pragma unroll
    for (int j = 10; j < 32; ++j) row[j] = (_Float16)0.0f;
    const float4* src = (const float4*)(agg1 + (size_t)(b * NPTS + idn) * D2);
    _Float16* dst = catm + t * 128;
#pragma unroll
    for (int c4 = 0; c4 < 16; ++c4) {
      float4 v = src[c4];
      dst[c4 * 4 + 0] = (_Float16)v.x;
      dst[c4 * 4 + 1] = (_Float16)v.y;
      dst[c4 * 4 + 2] = (_Float16)v.z;
      dst[c4 * 4 + 3] = (_Float16)v.w;
    }
  }
  __syncthreads();

  // bb1 (recompute): 10->64 -> fxyz, BN+ReLU
  {
    const int nt = wv & 3, ob = nt * 16, o = ob + col;
    const v16h bfb = load_b_pad(bb1w, 10, 10, ob, 0, bb1g, BNINV, ln);
    const float bias = bb1b[o];
#pragma unroll
    for (int i = 0; i < 8; ++i) {
      int mt = (wv >> 2) + 2 * i;
      v8f acc = {};
      acc = wmma32(load_a_h(in10, 32, mt * 16, 0, ln), bfb, acc);
      _Float16* dst = fxyz + (mt * 16 + 8 * hi) * 64 + o;
#pragma unroll
      for (int j = 0; j < 8; ++j)
        dst[j * 64] = (_Float16)fmaxf(acc[j] + bias, 0.0f);
    }
  }
  __syncthreads();

  // bb2: 64->64 over f_xyz -> cat[:,64:128], BN+ReLU
  {
    const int nt = wv & 3, ob = nt * 16, o = ob + col;
    const v16h b0 = load_b_full(bb2w, 64, ob, 0, bb2g, BNINV, ln);
    const v16h b1 = load_b_full(bb2w, 64, ob, 32, bb2g, BNINV, ln);
    const float bias = bb2b[o];
#pragma unroll
    for (int i = 0; i < 8; ++i) {
      int mt = (wv >> 2) + 2 * i;
      v8f acc = {};
      acc = wmma32(load_a_h(fxyz, 64, mt * 16, 0, ln), b0, acc);
      acc = wmma32(load_a_h(fxyz, 64, mt * 16, 32, ln), b1, acc);
      _Float16* dst = catm + (mt * 16 + 8 * hi) * 128 + 64 + o;
#pragma unroll
      for (int j = 0; j < 8; ++j)
        dst[j * 128] = (_Float16)fmaxf(acc[j] + bias, 0.0f);
    }
  }
  __syncthreads();

  // att2 + fused softmax pool -> aggs[16][128]
  {
    const int ob = wv * 16;
    const v16h w0 = load_b_full(a2fc, 128, ob, 0, nullptr, 1.0f, ln);
    const v16h w1 = load_b_full(a2fc, 128, ob, 32, nullptr, 1.0f, ln);
    const v16h w2 = load_b_full(a2fc, 128, ob, 64, nullptr, 1.0f, ln);
    const v16h w3 = load_b_full(a2fc, 128, ob, 96, nullptr, 1.0f, ln);
    float* sc = atts + wv * 256;
#pragma unroll 1
    for (int p = 0; p < 16; ++p) {
      v8f acc = {};
      acc = wmma32(load_a_h(catm, 128, p * 16, 0, ln), w0, acc);
      acc = wmma32(load_a_h(catm, 128, p * 16, 32, ln), w1, acc);
      acc = wmma32(load_a_h(catm, 128, p * 16, 64, ln), w2, acc);
      acc = wmma32(load_a_h(catm, 128, p * 16, 96, ln), w3, acc);
#pragma unroll
      for (int j = 0; j < 8; ++j) sc[(j + 8 * hi) * 16 + col] = acc[j];
      wave_lds_fence();
      if (ln < 16) {
        int o = ob + ln;
        float m = -1e30f;
#pragma unroll
        for (int k = 0; k < 16; ++k) m = fmaxf(m, sc[k * 16 + ln]);
        float s = 0.f, tt = 0.f;
#pragma unroll
        for (int k = 0; k < 16; ++k) {
          float e = __expf(sc[k * 16 + ln] - m);
          s += e;
          tt += e * (float)catm[(p * 16 + k) * 128 + o];
        }
        aggs[p * 128 + o] = tt / s;
      }
      wave_lds_fence();
    }
  }
  __syncthreads();

  // a2: 128 -> 128, BN+ReLU -> f2h
  {
    int ob = wv * 16, o = ob + col;
    v8f acc = {};
#pragma unroll
    for (int kc = 0; kc < 128; kc += 32) {
      v16h af = load_a_f(aggs, 128, 0, kc, ln);
      v16h bf = load_b_full(a2w, 128, ob, kc, a2g, BNINV, ln);
      acc = wmma32(af, bf, acc);
    }
    float bias = a2b[o];
    _Float16* dst = f2h + (8 * hi) * 128 + o;
#pragma unroll
    for (int j = 0; j < 8; ++j)
      dst[j * 128] = (_Float16)fmaxf(acc[j] + bias, 0.0f);
  }
  __syncthreads();

  // m2 (128->256) + shortcut (64->256) fused in one accumulator, leaky-relu
#pragma unroll
  for (int it = 0; it < 2; ++it) {
    int nt = wv + 8 * it, ob = nt * 16, o = ob + col;
    v8f acc = {};
#pragma unroll
    for (int kc = 0; kc < 128; kc += 32) {
      v16h af = load_a_h(f2h, 128, 0, kc, ln);
      v16h bf = load_b_full(m2w, 128, ob, kc, m2g, BNINV, ln);
      acc = wmma32(af, bf, acc);
    }
#pragma unroll
    for (int kc = 0; kc < 64; kc += 32) {
      v16h af = load_a_h(finh, 64, 0, kc, ln);
      v16h bf = load_b_full(scw, 64, ob, kc, scg, BNINV, ln);
      acc = wmma32(af, bf, acc);
    }
    float bias = m2b[o] + scb[o];
    float* dst = out + (size_t)(b * DO2 + o) * NPTS + n0 + 8 * hi;
#pragma unroll
    for (int j = 0; j < 8; ++j) {
      float v = acc[j] + bias;
      dst[j] = (v > 0.0f) ? v : 0.2f * v;
    }
  }
}

// ============================================================================
extern "C" void kernel_launch(void* const* d_in, const int* in_sizes, int n_in,
                              void* d_out, int out_size, void* d_ws, size_t ws_size,
                              hipStream_t stream) {
  (void)in_sizes; (void)n_in; (void)out_size; (void)ws_size;
  const float* feature = (const float*)d_in[0];
  const float* xyz     = (const float*)d_in[1];
  const int*   nidx    = (const int*)d_in[2];
  const float* m1w = (const float*)d_in[3];
  const float* m1g = (const float*)d_in[4];
  const float* m1b = (const float*)d_in[5];
  const float* bb1w = (const float*)d_in[6];
  const float* bb1g = (const float*)d_in[7];
  const float* bb1b = (const float*)d_in[8];
  const float* a1fc = (const float*)d_in[9];
  const float* a1w = (const float*)d_in[10];
  const float* a1g = (const float*)d_in[11];
  const float* a1b = (const float*)d_in[12];
  const float* bb2w = (const float*)d_in[13];
  const float* bb2g = (const float*)d_in[14];
  const float* bb2b = (const float*)d_in[15];
  const float* a2fc = (const float*)d_in[16];
  const float* a2w = (const float*)d_in[17];
  const float* a2g = (const float*)d_in[18];
  const float* a2b = (const float*)d_in[19];
  const float* m2w = (const float*)d_in[20];
  const float* m2g = (const float*)d_in[21];
  const float* m2b = (const float*)d_in[22];
  const float* scw = (const float*)d_in[23];
  const float* scg = (const float*)d_in[24];
  const float* scb = (const float*)d_in[25];
  float* outp = (float*)d_out;

  float* f1   = (float*)d_ws;                        // [B,N,64] f32
  float* agg1 = f1 + (size_t)BATCH * NPTS * D2;      // [B,N,64] f32

  k_mlp1<<<BATCH * (NPTS / 128), 256, 0, stream>>>(feature, m1w, m1g, m1b, f1);
  k_block1<<<BATCH * (NPTS / 16), 256, 99584, stream>>>(
      xyz, nidx, f1, bb1w, bb1g, bb1b, a1fc, a1w, a1g, a1b, agg1);
  k_block2<<<BATCH * (NPTS / 16), 256, 138496, stream>>>(
      feature, xyz, nidx, agg1, bb1w, bb1g, bb1b, bb2w, bb2g, bb2b,
      a2fc, a2w, a2g, a2b, m2w, m2g, m2b, scw, scg, scb, outp);
}
